// OutputLayer_21380347200142
// MI455X (gfx1250) — matrix operand order repaired and checked by reading, not compiled, not run on hardware
//
#include <hip/hip_runtime.h>
#include <hip/hip_bf16.h>
#include <math.h>

// Problem constants (match reference)
#define BDIM 8
#define LDIM 2500
#define DDIM 512
#define YDIM 8921

#define WAVES_PER_WG 8
#define YT_PER_WAVE 16
#define YT_PER_WG (WAVES_PER_WG * YT_PER_WAVE)   // 128
#define KFRAGS (DDIM / 32)                       // 16 wmma k-steps over D
#define NCHUNK 32                                // l-columns per iteration (2 N-tiles)

typedef __attribute__((ext_vector_type(16))) __bf16        v16bf;
typedef __attribute__((ext_vector_type(16))) short         v16s;
typedef __attribute__((ext_vector_type(8)))  float         v8f;
typedef __attribute__((ext_vector_type(8)))  unsigned int  v8u;

__device__ __forceinline__ unsigned int fbits(float f) {
    return __builtin_bit_cast(unsigned int, f);
}
// pack bf16(a) | bf16(b)<<16 via one v_perm_b32 (truncation rounding)
__device__ __forceinline__ unsigned int pack_bf16x2(float a, float b) {
    return __byte_perm(fbits(a), fbits(b), 0x7632);
}

// Fused flash-attention-style kernel:
//   S = U @ x^T , W = F @ x^T  (shared B operand), per-lane online softmax
//   over L with end-of-loop cross-lane merge,
//   y = (sum e^{S-max} * W) / (sum e^{S-max}) + bias
__global__ __launch_bounds__(256, 1)
void attn_pool_wmma_kernel(const float* __restrict__ x,
                           const float* __restrict__ Uw,
                           const float* __restrict__ Fw,
                           const float* __restrict__ bias,
                           float* __restrict__ y_out)
{
    // double-buffered bf16 staging: 2 x 32 KB (WGP LDS is 320 KB)
    __shared__ unsigned short lds_x[2][NCHUNK * DDIM];

    const int tid     = threadIdx.x;
    const int wave    = tid >> 5;
    const int lane    = tid & 31;
    const int half_id = lane >> 4;    // 0: lanes 0-15, 1: lanes 16-31
    const int lane16  = lane & 15;

    const int b      = blockIdx.x % BDIM;
    const int ytile0 = (blockIdx.x / BDIM) * YT_PER_WG + wave * YT_PER_WAVE;

    const float* __restrict__ xb = x + (size_t)b * LDIM * DDIM;

    // ---- Resident A-fragments for U and F (bf16), 16x32 per fragment ----
    // A order used here (contiguous k, same as B): lanes 0-15 -> M=lane, lanes 16-31 -> M=lane-16;
    // half element e maps to K = Kbase + e, Kbase = half_id*16.
    // Elements (2j, 2j+1) are d-contiguous -> one v_perm_b32 per dword.
    v16bf aU[KFRAGS], aF[KFRAGS];
    {
        int row = ytile0 + lane16;
        if (row >= YDIM) row = YDIM - 1;            // clamp; masked at store
        const float* urow = Uw + (size_t)row * DDIM;
        const float* frow = Fw + (size_t)row * DDIM;
        const int kb = half_id * 16;
#pragma unroll
        for (int kf = 0; kf < KFRAGS; ++kf) {
            const int d0 = kf * 32 + kb;
            v8u pu, pf;
#pragma unroll
            for (int j = 0; j < 4; ++j) {           // halfs 0..7 : d = d0+2j
                pu[j] = pack_bf16x2(urow[d0 + 2 * j], urow[d0 + 2 * j + 1]);
                pf[j] = pack_bf16x2(frow[d0 + 2 * j], frow[d0 + 2 * j + 1]);
            }
#pragma unroll
            for (int j = 0; j < 4; ++j) {           // halfs 8..15: d = d0+8+2j
                pu[4 + j] = pack_bf16x2(urow[d0 + 8 + 2 * j], urow[d0 + 9 + 2 * j]);
                pf[4 + j] = pack_bf16x2(frow[d0 + 8 + 2 * j], frow[d0 + 9 + 2 * j]);
            }
            aU[kf] = __builtin_bit_cast(v16bf, pu);
            aF[kf] = __builtin_bit_cast(v16bf, pf);
        }
    }

    // per-lane online-softmax state; slot r holds row (r + half_id*8),
    // columns owned by this lane: l = chunk*32 + t*16 + lane16
    float lmax[8], lden[8], lnum[8];
#pragma unroll
    for (int r = 0; r < 8; ++r) { lmax[r] = -INFINITY; lden[r] = 0.f; lnum[r] = 0.f; }

    const int nchunks = (LDIM + NCHUNK - 1) / NCHUNK;   // 79

    // cooperative stage of one chunk into lds buffer `buf`:
    // 32x512 f32 -> bf16; each thread handles 8 groups of 8 elements.
    auto stage = [&](int c, int buf) {
        const int l0 = c * NCHUNK;
#pragma unroll
        for (int i = 0; i < (NCHUNK * DDIM) / (256 * 8); ++i) {   // 8 iters
            const int idx  = (tid + i * 256) * 8;
            const int lrow = idx / DDIM;
            const int dcol = idx % DDIM;
            const int l    = l0 + lrow;
            float4 v0 = make_float4(0.f, 0.f, 0.f, 0.f);
            float4 v1 = make_float4(0.f, 0.f, 0.f, 0.f);
            if (l < LDIM) {
                const float* p = xb + (size_t)l * DDIM + dcol;
                v0 = *(const float4*)p;
                v1 = *(const float4*)(p + 4);
            }
            uint4 pk;
            pk.x = pack_bf16x2(v0.x, v0.y);
            pk.y = pack_bf16x2(v0.z, v0.w);
            pk.z = pack_bf16x2(v1.x, v1.y);
            pk.w = pack_bf16x2(v1.z, v1.w);
            *(uint4*)&lds_x[buf][idx] = pk;     // ds_store_b128
        }
    };

    stage(0, 0);
    __syncthreads();

    for (int c = 0; c < nchunks; ++c) {
        // stage next chunk into the other buffer (overlaps this chunk's math)
        if (c + 1 < nchunks) stage(c + 1, (c + 1) & 1);
        // prefetch chunk c+2 toward the WGP (speculative; OOB silently dropped)
        if (c + 2 < nchunks) {
            const char* pf = (const char*)(xb + (size_t)(c + 2) * NCHUNK * DDIM);
            __builtin_prefetch(pf + tid * 128, 0, 1);
            __builtin_prefetch(pf + 32768 + tid * 128, 0, 1);
        }

        // ---- 4 independent accumulation chains: 64 WMMAs per chunk ----
        // B order used here (32x16, contiguous k as in A): lanes 0-15 hold K=0..15, lanes 16-31 hold
        // K=16..31; element e is K = half_id*16 + e, N = lane16.
        const unsigned short* lx = lds_x[c & 1];
        auto loadB = [&](int kf, int t) -> v16bf {
            return __builtin_bit_cast(
                v16bf, *(const v16s*)&lx[(t * 16 + lane16) * DDIM
                                         + kf * 32 + half_id * 16]);
        };

        v8f S0 = {}, S1 = {}, W0 = {}, W1 = {};
        // 1-deep software pipeline: next k-step's B loads issue before this
        // step's 4 WMMAs, giving each ds_load 4 WMMAs of latency cover.
        v16bf b0 = loadB(0, 0), b1 = loadB(0, 1);
#pragma unroll
        for (int kf = 0; kf < KFRAGS; ++kf) {
            const int kn = (kf + 1 < KFRAGS) ? kf + 1 : kf;  // last pair is DCE'd
            const v16bf nb0 = loadB(kn, 0);
            const v16bf nb1 = loadB(kn, 1);
            S0 = __builtin_amdgcn_wmma_f32_16x16x32_bf16(false, aU[kf], false, b0,
                                                         (short)0, S0, false, false);
            W0 = __builtin_amdgcn_wmma_f32_16x16x32_bf16(false, aF[kf], false, b0,
                                                         (short)0, W0, false, false);
            S1 = __builtin_amdgcn_wmma_f32_16x16x32_bf16(false, aU[kf], false, b1,
                                                         (short)0, S1, false, false);
            W1 = __builtin_amdgcn_wmma_f32_16x16x32_bf16(false, aF[kf], false, b1,
                                                         (short)0, W1, false, false);
            b0 = nb0; b1 = nb1;
        }

        // ---- per-lane online softmax update (no cross-lane ops in loop) ----
        const int l0 = c * NCHUNK;
        const bool v0c = (l0 + lane16) < LDIM;
        const bool v1c = (l0 + 16 + lane16) < LDIM;
#pragma unroll
        for (int r = 0; r < 8; ++r) {
            const float s0 = v0c ? S0[r] : -INFINITY;
            const float s1 = v1c ? S1[r] : -INFINITY;
            const float w0 = v0c ? W0[r] : 0.f;
            const float w1 = v1c ? W1[r] : 0.f;
            const float nm = fmaxf(lmax[r], fmaxf(s0, s1));
            const float sc = __expf(lmax[r] - nm);     // 0 on first chunk
            const float p0 = __expf(s0 - nm);          // 0 for masked cols
            const float p1 = __expf(s1 - nm);
            lden[r] = lden[r] * sc + (p0 + p1);
            lnum[r] = lnum[r] * sc + (p0 * w0 + p1 * w1);
            lmax[r] = nm;
        }

        __syncthreads();   // buffer (c+1)&1 fully staged before next compute
    }

    // ---- merge 16 per-lane softmax states per half (once) ----
#pragma unroll
    for (int r = 0; r < 8; ++r) {
        float m = lmax[r], d = lden[r], n = lnum[r];
#pragma unroll
        for (int off = 1; off < 16; off <<= 1) {   // stays within the half
            const float om = __shfl_xor(m, off);
            const float od = __shfl_xor(d, off);
            const float on = __shfl_xor(n, off);
            const float nm = fmaxf(m, om);
            const float sa = __expf(m - nm);
            const float sb = __expf(om - nm);
            d = d * sa + od * sb;
            n = n * sa + on * sb;
            m = nm;
        }
        lden[r] = d; lnum[r] = n;
    }

    if (lane16 == 0) {
#pragma unroll
        for (int r = 0; r < 8; ++r) {
            const int row = ytile0 + r + half_id * 8;
            if (row < YDIM)
                y_out[(size_t)b * YDIM + row] = lnum[r] / lden[r] + bias[row];
        }
    }
}

// Mean BCE-with-logits over all B*Y logits (single workgroup).
__global__ __launch_bounds__(256)
void bce_loss_kernel(const float* __restrict__ y,
                     const float* __restrict__ target,
                     float* __restrict__ loss_out)
{
    __shared__ float red[256];
    float acc = 0.f;
    for (int i = threadIdx.x; i < BDIM * YDIM; i += 256) {
        const float v = y[i], t = target[i];
        acc += fmaxf(v, 0.f) - v * t + log1pf(__expf(-fabsf(v)));
    }
    red[threadIdx.x] = acc;
    __syncthreads();
    for (int s = 128; s > 0; s >>= 1) {
        if (threadIdx.x < s) red[threadIdx.x] += red[threadIdx.x + s];
        __syncthreads();
    }
    if (threadIdx.x == 0)
        loss_out[0] = red[0] / (float)(BDIM * YDIM);
}

extern "C" void kernel_launch(void* const* d_in, const int* in_sizes, int n_in,
                              void* d_out, int out_size, void* d_ws, size_t ws_size,
                              hipStream_t stream) {
    (void)in_sizes; (void)n_in; (void)d_ws; (void)ws_size; (void)out_size;
    const float* x      = (const float*)d_in[0];
    const float* target = (const float*)d_in[1];
    // d_in[2] text_inputs: unused by the reference computation
    const float* Uw     = (const float*)d_in[3];
    const float* Fw     = (const float*)d_in[4];
    const float* bias   = (const float*)d_in[5];

    float* y_out = (float*)d_out;                  // [B*Y]
    float* loss  = y_out + (size_t)BDIM * YDIM;    // [1]

    const int ytiles = (YDIM + YT_PER_WG - 1) / YT_PER_WG;   // 70
    dim3 grid(BDIM * ytiles);                                // 560 WGs
    attn_pool_wmma_kernel<<<grid, 256, 0, stream>>>(x, Uw, Fw, bias, y_out);
    bce_loss_kernel<<<1, 256, 0, stream>>>(y_out, target, loss);
}
